// Decoder_10539849744453
// MI455X (gfx1250) — matrix operand order repaired and checked by reading, not compiled, not run on hardware
//
#include <hip/hip_runtime.h>

#define VOCABN 10000
#define EMB 512
#define HID 1024
#define KSZ 128
#define VSZ 128
#define NB 64
#define T_ENC 500
#define T_DEC 200
#define K1 (EMB + VSZ + HID)   /* 1664 */
#define K2 (HID + KSZ)         /* 1152 */
#define K3 (KSZ + VSZ)         /* 256  */

typedef __bf16 bf16_t;
typedef __attribute__((ext_vector_type(16))) __bf16 v16bf;
typedef __attribute__((ext_vector_type(8)))  __bf16 v8bf;
typedef __attribute__((ext_vector_type(8)))  float  v8f;

__device__ __forceinline__ float sigmoidf(float x) { return 1.0f / (1.0f + __expf(-x)); }

// 16x32 bf16 A-fragment (row-major activation, leading dim ld).
// Lane L: row = row0 + (L&15); K contiguous {kb..kb+15}, kb = k0 + (L>>4)*16.
__device__ __forceinline__ v16bf load_a(const bf16_t* base, int ld, int row, int k0, int lane) {
    int g = lane >> 4;
    const bf16_t* p = base + (size_t)row * ld + k0 + g * 16;
    v8bf lo = *(const v8bf*)p;
    v8bf hi = *(const v8bf*)(p + 8);
    return __builtin_shufflevector(lo, hi, 0,1,2,3,4,5,6,7,8,9,10,11,12,13,14,15);
}

// One K-chunk worth of operands: B fragment (32x16 of W^T) + A fragments for all 4 M-tiles.
struct Frag { v16bf b, a0, a1, a2, a3; };

__device__ __forceinline__ void load_frags(Frag& f, const bf16_t* wrow, const bf16_t* act,
                                           int ld, int nl, int lane, int k0) {
    f.b  = *(const v16bf*)(wrow + k0);
    f.a0 = load_a(act, ld,  0 + nl, k0, lane);
    f.a1 = load_a(act, ld, 16 + nl, k0, lane);
    f.a2 = load_a(act, ld, 32 + nl, k0, lane);
    f.a3 = load_a(act, ld, 48 + nl, k0, lane);
}

__device__ __forceinline__ void wmma4(v8f acc[4], const Frag& f) {
    acc[0] = __builtin_amdgcn_wmma_f32_16x16x32_bf16(false, f.a0, false, f.b, (short)0, acc[0], false, false);
    acc[1] = __builtin_amdgcn_wmma_f32_16x16x32_bf16(false, f.a1, false, f.b, (short)0, acc[1], false, false);
    acc[2] = __builtin_amdgcn_wmma_f32_16x16x32_bf16(false, f.a2, false, f.b, (short)0, acc[2], false, false);
    acc[3] = __builtin_amdgcn_wmma_f32_16x16x32_bf16(false, f.a3, false, f.b, (short)0, acc[3], false, false);
}

// Ping-pong double-buffered K loop (K divisible by 64): next chunk's loads are
// in flight before the current chunk's WMMAs, so waits are partial, not full.
__device__ __forceinline__ void gemm_k_loop(v8f acc[4], const bf16_t* wrow, const bf16_t* act,
                                            int K, int nl, int lane) {
    Frag f0, f1;
    load_frags(f0, wrow, act, K, nl, lane, 0);
    int k0;
    for (k0 = 0; k0 + 64 < K; k0 += 64) {
        __builtin_prefetch(wrow + k0 + 256, 0, 1);   // stays inside contiguous workspace
        load_frags(f1, wrow, act, K, nl, lane, k0 + 32);
        wmma4(acc, f0);
        load_frags(f0, wrow, act, K, nl, lane, k0 + 64);
        wmma4(acc, f1);
    }
    load_frags(f1, wrow, act, K, nl, lane, K - 32);
    wmma4(acc, f0);
    wmma4(acc, f1);
}

// ---------------- one-time prep: fp32 -> bf16 weight packing ----------------
__global__ void k_prep_w1(const float* Wih, const float* Whh, const float* bih, const float* bhh,
                          bf16_t* W1, float* b1) {
    int j = blockIdx.x, tid = threadIdx.x;
    for (int k = tid; k < EMB + VSZ; k += 256)
        W1[(size_t)j * K1 + k] = (bf16_t)Wih[(size_t)j * (EMB + VSZ) + k];
    for (int k = tid; k < HID; k += 256)
        W1[(size_t)j * K1 + (EMB + VSZ) + k] = (bf16_t)Whh[(size_t)j * HID + k];
    if (tid == 0) b1[j] = bih[j] + bhh[j];
}

__global__ void k_prep_w2(const float* Wih, const float* Whh, const float* bih, const float* bhh,
                          bf16_t* W2, float* b2) {
    int j = blockIdx.x, tid = threadIdx.x;
    for (int k = tid; k < HID; k += 256)
        W2[(size_t)j * K2 + k] = (bf16_t)Wih[(size_t)j * HID + k];
    for (int k = tid; k < KSZ; k += 256)
        W2[(size_t)j * K2 + HID + k] = (bf16_t)Whh[(size_t)j * KSZ + k];
    if (tid == 0) b2[j] = bih[j] + bhh[j];
}

__global__ void k_prep_wout(const float* Wout_f, bf16_t* Wout) {
    int j = blockIdx.x, tid = threadIdx.x;
    if (tid < K3) Wout[(size_t)j * K3 + tid] = (bf16_t)Wout_f[(size_t)j * K3 + tid];
}

__global__ void k_init(float* c1, float* c2, float* h2f, bf16_t* h1bf, bf16_t* h2bf) {
    int i = blockIdx.x * blockDim.x + threadIdx.x;
    if (i < NB * HID) { c1[i] = 0.f; h1bf[i] = (bf16_t)0.f; }
    if (i < NB * KSZ) { c2[i] = 0.f; h2f[i] = 0.f; h2bf[i] = (bf16_t)0.f; }
}

// ---------------- per-step: attention + activation packing ----------------
__global__ void k_att(const float* __restrict__ key, const float* __restrict__ value,
                      const int* __restrict__ src_lens, const int* __restrict__ text,
                      const float* __restrict__ emb_table, int t,
                      const float* __restrict__ h2f, const bf16_t* __restrict__ h1bf,
                      const bf16_t* __restrict__ h2bf,
                      bf16_t* __restrict__ x1, bf16_t* __restrict__ x2, bf16_t* __restrict__ x3) {
    __shared__ float sh2[KSZ];
    __shared__ float se[T_ENC];
    __shared__ float sred[256];
    int n = blockIdx.x, tid = threadIdx.x;
    for (int j = tid; j < KSZ; j += 256) sh2[j] = h2f[n * KSZ + j];
    __syncthreads();
    int slen = src_lens[n];
    float lmax = -3.4e38f;
    for (int tt = tid; tt < T_ENC; tt += 256) {
        float e;
        if (tt >= slen) e = -1e9f;
        else {
            const float* kp = key + ((size_t)n * T_ENC + tt) * KSZ;
            e = 0.f;
            #pragma unroll 8
            for (int j = 0; j < KSZ; j++) e += kp[j] * sh2[j];
        }
        se[tt] = e;
        lmax = fmaxf(lmax, e);
    }
    sred[tid] = lmax; __syncthreads();
    for (int s = 128; s > 0; s >>= 1) { if (tid < s) sred[tid] = fmaxf(sred[tid], sred[tid + s]); __syncthreads(); }
    float mx = sred[0]; __syncthreads();
    float lsum = 0.f;
    for (int tt = tid; tt < T_ENC; tt += 256) { float p = __expf(se[tt] - mx); se[tt] = p; lsum += p; }
    sred[tid] = lsum; __syncthreads();
    for (int s = 128; s > 0; s >>= 1) { if (tid < s) sred[tid] += sred[tid + s]; __syncthreads(); }
    float inv = 1.0f / sred[0];
    if (tid < VSZ) {
        const float* vp = value + (size_t)n * T_ENC * VSZ + tid;
        float acc = 0.f;
        for (int tt = 0; tt < T_ENC; tt++) acc += se[tt] * vp[(size_t)tt * VSZ];
        bf16_t cb = (bf16_t)(acc * inv);
        x1[(size_t)n * K1 + EMB + tid] = cb;
        x3[n * K3 + KSZ + tid] = cb;
    }
    int tok = text[n * T_DEC + t];
    const float* ep = emb_table + (size_t)tok * EMB;
    for (int j = tid; j < EMB; j += 256) x1[(size_t)n * K1 + j] = (bf16_t)ep[j];
    for (int j = tid; j < HID; j += 256) x1[(size_t)n * K1 + EMB + VSZ + j] = h1bf[n * HID + j];
    for (int j = tid; j < KSZ; j += 256) x2[n * K2 + HID + j] = h2bf[n * KSZ + j];
}

// ---------------- per-step: LSTM1 gates GEMM + fused cell update ----------------
// 64 blocks x 128 threads (4 waves). Wave q owns gate q for all 4 M-tiles:
// W1 is streamed exactly once per step. __launch_bounds__(128,1): one wave per
// SIMD is the natural occupancy here, so let the allocator use the full VGPR
// file and keep both fragment buffers + 4 accumulators resident (no spills).
__global__ void __launch_bounds__(128, 1)
k_gemm1(const bf16_t* __restrict__ x1, const bf16_t* __restrict__ W1,
        const float* __restrict__ b1,
        float* __restrict__ c1, bf16_t* __restrict__ h1bf, bf16_t* __restrict__ x2) {
    __shared__ float sg[4][NB][16];   // 16 KB
    int tid = threadIdx.x;
    int lane = tid & 31;
    int q = tid >> 5;
    int j0 = blockIdx.x * 16;
    int nl = lane & 15, g = lane >> 4;
    float b = b1[q * HID + j0 + nl];
    v8f acc[4];
    #pragma unroll
    for (int mt = 0; mt < 4; mt++) { v8f tmp = {b, b, b, b, b, b, b, b}; acc[mt] = tmp; }
    const bf16_t* wrow = W1 + (size_t)(q * HID + j0 + nl) * K1 + g * 16;
    gemm_k_loop(acc, wrow, x1, K1, nl, lane);
    #pragma unroll
    for (int mt = 0; mt < 4; mt++)
        #pragma unroll
        for (int v = 0; v < 8; v++)
            sg[q][mt * 16 + g * 8 + v][nl] = acc[mt][v];
    __syncthreads();
    #pragma unroll
    for (int i = 0; i < 8; i++) {
        int e = tid * 8 + i;
        int m = e >> 4, jj = e & 15;
        int jc = j0 + jj;
        float c_old = c1[m * HID + jc];
        float cn = sigmoidf(sg[1][m][jj]) * c_old + sigmoidf(sg[0][m][jj]) * tanhf(sg[2][m][jj]);
        float hn = sigmoidf(sg[3][m][jj]) * tanhf(cn);
        c1[m * HID + jc] = cn;
        bf16_t hb = (bf16_t)hn;
        h1bf[m * HID + jc] = hb;       // staged for x1 repack (next step's k_att)
        x2[m * K2 + jc] = hb;          // safe: k_gemm2 launches after us
    }
}

// ---------------- per-step: LSTM2 gates GEMM + fused cell update ----------------
__global__ void __launch_bounds__(128, 1)
k_gemm2(const bf16_t* __restrict__ x2, const bf16_t* __restrict__ W2,
        const float* __restrict__ b2,
        float* __restrict__ c2, float* __restrict__ h2f,
        bf16_t* __restrict__ h2bf, bf16_t* __restrict__ x3) {
    __shared__ float sg[4][NB][16];
    int tid = threadIdx.x;
    int lane = tid & 31;
    int q = tid >> 5;
    int j0 = blockIdx.x * 16;
    int nl = lane & 15, g = lane >> 4;
    float b = b2[q * KSZ + j0 + nl];
    v8f acc[4];
    #pragma unroll
    for (int mt = 0; mt < 4; mt++) { v8f tmp = {b, b, b, b, b, b, b, b}; acc[mt] = tmp; }
    const bf16_t* wrow = W2 + (size_t)(q * KSZ + j0 + nl) * K2 + g * 16;
    gemm_k_loop(acc, wrow, x2, K2, nl, lane);
    #pragma unroll
    for (int mt = 0; mt < 4; mt++)
        #pragma unroll
        for (int v = 0; v < 8; v++)
            sg[q][mt * 16 + g * 8 + v][nl] = acc[mt][v];
    __syncthreads();
    #pragma unroll
    for (int i = 0; i < 8; i++) {
        int e = tid * 8 + i;
        int m = e >> 4, jj = e & 15;
        int jc = j0 + jj;
        float c_old = c2[m * KSZ + jc];
        float cn = sigmoidf(sg[1][m][jj]) * c_old + sigmoidf(sg[0][m][jj]) * tanhf(sg[2][m][jj]);
        float hn = sigmoidf(sg[3][m][jj]) * tanhf(cn);
        c2[m * KSZ + jc] = cn;
        h2f[m * KSZ + jc] = hn;        // fp32 for next step's attention energies
        bf16_t hb = (bf16_t)hn;
        h2bf[m * KSZ + jc] = hb;       // staged for x2 repack (next step's k_att)
        x3[m * K3 + jc] = hb;          // safe: k_logits launches after us
    }
}

// ---------------- per-step: output projection [h2|ctx] @ W_out^T + b ----------------
// 625 blocks x 32 threads (ONE wave per vocab tile, all 4 M-tiles):
// W_out streamed exactly once per step.
__global__ void __launch_bounds__(32, 1)
k_logits(const bf16_t* __restrict__ x3, const bf16_t* __restrict__ Wout,
         const float* __restrict__ b_out, float* __restrict__ out, int t) {
    int lane = threadIdx.x & 31;
    int v0 = blockIdx.x * 16;
    int nl = lane & 15, g = lane >> 4;
    float b = b_out[v0 + nl];
    v8f acc[4];
    #pragma unroll
    for (int mt = 0; mt < 4; mt++) { v8f tmp = {b, b, b, b, b, b, b, b}; acc[mt] = tmp; }
    const bf16_t* wrow = Wout + (size_t)(v0 + nl) * K3 + g * 16;
    gemm_k_loop(acc, wrow, x3, K3, nl, lane);
    #pragma unroll
    for (int mt = 0; mt < 4; mt++)
        #pragma unroll
        for (int v = 0; v < 8; v++) {
            int m = mt * 16 + g * 8 + v;
            out[((size_t)m * T_DEC + t) * VOCABN + v0 + nl] = acc[mt][v];
        }
}

extern "C" void kernel_launch(void* const* d_in, const int* in_sizes, int n_in,
                              void* d_out, int out_size, void* d_ws, size_t ws_size,
                              hipStream_t stream) {
    (void)in_sizes; (void)n_in; (void)out_size; (void)ws_size;
    const float* key       = (const float*)d_in[0];
    const float* value     = (const float*)d_in[1];
    const int*   src_lens  = (const int*)  d_in[2];
    const int*   text      = (const int*)  d_in[3];
    const float* emb_table = (const float*)d_in[4];
    const float* W_ih1     = (const float*)d_in[5];
    const float* W_hh1     = (const float*)d_in[6];
    const float* b_ih1     = (const float*)d_in[7];
    const float* b_hh1     = (const float*)d_in[8];
    const float* W_ih2     = (const float*)d_in[9];
    const float* W_hh2     = (const float*)d_in[10];
    const float* b_ih2     = (const float*)d_in[11];
    const float* b_hh2     = (const float*)d_in[12];
    const float* W_out     = (const float*)d_in[13];
    const float* b_out     = (const float*)d_in[14];
    float* out = (float*)d_out;

    char* ws = (char*)d_ws;
    size_t off = 0;
    auto alloc = [&](size_t bytes) -> void* {
        void* p = ws + off;
        off = (off + bytes + 255) & ~(size_t)255;
        return p;
    };
    bf16_t* W1   = (bf16_t*)alloc((size_t)4 * HID * K1 * sizeof(bf16_t));   // 13.6 MB
    float*  b1   = (float*) alloc((size_t)4 * HID * sizeof(float));
    bf16_t* W2   = (bf16_t*)alloc((size_t)4 * KSZ * K2 * sizeof(bf16_t));   // 1.2 MB
    float*  b2   = (float*) alloc((size_t)4 * KSZ * sizeof(float));
    bf16_t* Wout = (bf16_t*)alloc((size_t)VOCABN * K3 * sizeof(bf16_t));    // 5.1 MB
    bf16_t* x1   = (bf16_t*)alloc((size_t)NB * K1 * sizeof(bf16_t));
    bf16_t* x2   = (bf16_t*)alloc((size_t)NB * K2 * sizeof(bf16_t));
    bf16_t* x3   = (bf16_t*)alloc((size_t)NB * K3 * sizeof(bf16_t));
    float*  c1   = (float*) alloc((size_t)NB * HID * sizeof(float));
    float*  c2   = (float*) alloc((size_t)NB * KSZ * sizeof(float));
    float*  h2f  = (float*) alloc((size_t)NB * KSZ * sizeof(float));
    bf16_t* h1bf = (bf16_t*)alloc((size_t)NB * HID * sizeof(bf16_t));
    bf16_t* h2bf = (bf16_t*)alloc((size_t)NB * KSZ * sizeof(bf16_t));

    // one-time (per launch) prep: bf16 weight packing + zero state
    k_prep_w1  <<<4 * HID, 256, 0, stream>>>(W_ih1, W_hh1, b_ih1, b_hh1, W1, b1);
    k_prep_w2  <<<4 * KSZ, 256, 0, stream>>>(W_ih2, W_hh2, b_ih2, b_hh2, W2, b2);
    k_prep_wout<<<VOCABN,  256, 0, stream>>>(W_out, Wout);
    k_init     <<<(NB * HID + 255) / 256, 256, 0, stream>>>(c1, c2, h2f, h1bf, h2bf);

    for (int t = 0; t < T_DEC; t++) {
        k_att   <<<NB,  256, 0, stream>>>(key, value, src_lens, text, emb_table, t,
                                          h2f, h1bf, h2bf, x1, x2, x3);
        k_gemm1 <<<HID / 16, 128, 0, stream>>>(x1, W1, b1, c1, h1bf, x2);
        k_gemm2 <<<KSZ / 16, 128, 0, stream>>>(x2, W2, b2, c2, h2f, h2bf, x3);
        k_logits<<<VOCABN / 16, 32, 0, stream>>>(x3, Wout, b_out, out, t);
    }
}
